// DualStreamDTIClassifier_72310069395914
// MI455X (gfx1250) — matrix operand order repaired and checked by reading, not compiled, not run on hardware
//
#include <hip/hip_runtime.h>
#include <cstddef>
#include <cstdint>

// ---------------- problem constants (match reference) ----------------
#define B_    16
#define LL_   128
#define LP_   2048
#define D_    512
#define H_    8
#define DH_   64
#define DFF_  2048
#define NL_   4
#define ML_   (B_*LL_)    // 2048 l-stream rows
#define MP_   (B_*LP_)    // 32768 p-stream rows
#define SCALE_ 0.125f     // 1/sqrt(DH)
#define EPS_SM 0.02f

typedef __attribute__((ext_vector_type(16))) __bf16 v16bf;
typedef __attribute__((ext_vector_type(8)))  float  v8f;

enum { EPI_F32 = 0, EPI_BF16 = 1, EPI_GELU = 2, EPI_VT = 3, EPI_RESID = 4 };

// ---------------- fragment loaders (CDNA5 16-bit WMMA layouts) ----------------
// A 16x32 (MxK): lane l -> row M=l&15, half=l>>4.
//   VGPR0-3 : K = half*8 + [0..7]   (contiguous 16B)
//   VGPR4-7 : K = 16 + half*8 + [0..7] (contiguous 16B, +32B from first chunk)
__device__ __forceinline__ v16bf load_frag_a(const __bf16* row, int kk, int half) {
  union { uint4 u[2]; v16bf v; } f;
  const uint4* p = reinterpret_cast<const uint4*>(row + kk + half * 8);
  f.u[0] = p[0];
  f.u[1] = p[2];         // +16 elements = +32 bytes
  return f.v;
}
// B 32x16 (KxN): lane l -> col N=l&15; same K layout as A:
// K = half*8 + [0..7], then K = 16 + half*8 + [0..7].
__device__ __forceinline__ v16bf load_frag_b(const __bf16* row, int kk, int half) {
  union { uint4 u[2]; v16bf v; } f;
  const uint4* p = reinterpret_cast<const uint4*>(row + kk + half * 8);
  f.u[0] = p[0];
  f.u[1] = p[2];
  return f.v;
}

// ---------------- epilogues ----------------
// C tile layout: lane = N (0..15, repeated), VGPR r -> M = r + (lane>=16 ? 8 : 0)
template <int EPI>
__device__ __forceinline__ void epi_store(v8f acc, int m0, int n0, int lane,
                                          const float* bias, char* Cout, size_t coff,
                                          int ldc, float alpha, int LkVT) {
  const int nc = lane & 15, half = lane >> 4;
  const int n = n0 + nc;
  const float bv = bias ? bias[n] : 0.0f;
#pragma unroll
  for (int r = 0; r < 8; ++r) {
    const int m = m0 + r + half * 8;
    float v = acc[r] * alpha + bv;
    if (EPI == EPI_F32) {
      reinterpret_cast<float*>(Cout)[coff + (size_t)m * ldc + n] = v;
    } else if (EPI == EPI_BF16) {
      reinterpret_cast<__bf16*>(Cout)[coff + (size_t)m * ldc + n] = (__bf16)v;
    } else if (EPI == EPI_GELU) {
      float g = 0.5f * v * (1.0f + erff(v * 0.70710678118f));   // exact gelu
      reinterpret_cast<__bf16*>(Cout)[coff + (size_t)m * ldc + n] = (__bf16)g;
    } else if (EPI == EPI_VT) {
      // write V transposed: vT[b][h][d][kpos], m = b*LkVT + kpos, n = h*64 + d
      const int bb = m / LkVT, kp = m % LkVT;
      const int hh = n >> 6, dd = n & 63;
      reinterpret_cast<__bf16*>(Cout)[(((size_t)(bb * H_ + hh)) * DH_ + dd) * LkVT + kp] = (__bf16)v;
    } else { // EPI_RESID: residual-accumulate into f32 output
      float* p = reinterpret_cast<float*>(Cout) + coff + (size_t)m * ldc + n;
      *p = *p + v;
    }
  }
}

// ---------------- generic NT GEMM:  C(+)= alpha * A[M,K] . W[N,K]^T + bias ----------------
// block = 128 threads = 4 waves stacked in M; wave tile 32x64:
//   per K-step: 2 A-frags + 4 B-frags (6 x 32B/lane) -> 8 WMMAs (1.33 WMMA per fragment).
// block tile 128x64; all M here are multiples of 128, all N multiples of 64.
// batched via blockIdx.z -> (zb,zh) with independent strides (per-(b,h) attention GEMMs).
template <int EPI>
__global__ __launch_bounds__(128)
void gemm_nt_kernel(const __bf16* __restrict__ A, const __bf16* __restrict__ W,
                    const float* __restrict__ bias, char* __restrict__ Cout,
                    int K, int lda, int ldb, int ldc,
                    long long sAb, long long sAh, long long sBb, long long sBh,
                    long long sCb, long long sCh, int Hdiv, float alpha, int LkVT) {
  const int z = blockIdx.z;
  const int zb = z / Hdiv, zh = z % Hdiv;
  A += zb * sAb + zh * sAh;
  W += zb * sBb + zh * sBh;
  const size_t coff = (size_t)(zb * sCb + zh * sCh);

  const int wave = threadIdx.x >> 5;
  const int lane = threadIdx.x & 31;
  const int m0 = blockIdx.y * 128 + wave * 32;
  const int n0 = blockIdx.x * 64;
  const int r = lane & 15, half = lane >> 4;

  const __bf16* a0p = A + (size_t)(m0 + r) * lda;
  const __bf16* a1p = A + (size_t)(m0 + 16 + r) * lda;
  const __bf16* bp0 = W + (size_t)(n0 + r) * ldb;
  const __bf16* bp1 = W + (size_t)(n0 + 16 + r) * ldb;
  const __bf16* bp2 = W + (size_t)(n0 + 32 + r) * ldb;
  const __bf16* bp3 = W + (size_t)(n0 + 48 + r) * ldb;

  v8f acc[2][4] = {};
  for (int kk = 0; kk < K; kk += 32) {
    v16bf a0 = load_frag_a(a0p, kk, half);
    v16bf a1 = load_frag_a(a1p, kk, half);
    v16bf b0 = load_frag_b(bp0, kk, half);
    v16bf b1 = load_frag_b(bp1, kk, half);
    v16bf b2 = load_frag_b(bp2, kk, half);
    v16bf b3 = load_frag_b(bp3, kk, half);
    if (kk + 32 < K) {                       // global_prefetch_b8 of next K slice
      __builtin_prefetch(a0p + kk + 32, 0, 3);
      __builtin_prefetch(bp0 + kk + 32, 0, 3);
    }
    acc[0][0] = __builtin_amdgcn_wmma_f32_16x16x32_bf16(false, a0, false, b0, (short)0, acc[0][0], false, false);
    acc[0][1] = __builtin_amdgcn_wmma_f32_16x16x32_bf16(false, a0, false, b1, (short)0, acc[0][1], false, false);
    acc[0][2] = __builtin_amdgcn_wmma_f32_16x16x32_bf16(false, a0, false, b2, (short)0, acc[0][2], false, false);
    acc[0][3] = __builtin_amdgcn_wmma_f32_16x16x32_bf16(false, a0, false, b3, (short)0, acc[0][3], false, false);
    acc[1][0] = __builtin_amdgcn_wmma_f32_16x16x32_bf16(false, a1, false, b0, (short)0, acc[1][0], false, false);
    acc[1][1] = __builtin_amdgcn_wmma_f32_16x16x32_bf16(false, a1, false, b1, (short)0, acc[1][1], false, false);
    acc[1][2] = __builtin_amdgcn_wmma_f32_16x16x32_bf16(false, a1, false, b2, (short)0, acc[1][2], false, false);
    acc[1][3] = __builtin_amdgcn_wmma_f32_16x16x32_bf16(false, a1, false, b3, (short)0, acc[1][3], false, false);
  }
#pragma unroll
  for (int mi = 0; mi < 2; ++mi)
#pragma unroll
    for (int ni = 0; ni < 4; ++ni)
      epi_store<EPI>(acc[mi][ni], m0 + mi * 16, n0 + ni * 16, lane, bias, Cout, coff, ldc, alpha, LkVT);
}

// ---------------- LayerNorm (f32 in -> bf16 out), one wave per 512-col row ----------------
__global__ __launch_bounds__(256)
void layernorm_kernel(const float* __restrict__ x, const float* __restrict__ g,
                      const float* __restrict__ bta, __bf16* __restrict__ y, int M) {
  const int row = blockIdx.x * 8 + (threadIdx.x >> 5);
  const int lane = threadIdx.x & 31;
  if (row >= M) return;
  const float* xr = x + (size_t)row * D_;
  float v[16];
  float s = 0.f;
#pragma unroll
  for (int i = 0; i < 16; ++i) { v[i] = xr[lane + 32 * i]; s += v[i]; }
#pragma unroll
  for (int o = 16; o > 0; o >>= 1) s += __shfl_xor(s, o, 32);
  const float mean = s * (1.0f / D_);
  float var = 0.f;
#pragma unroll
  for (int i = 0; i < 16; ++i) { float d = v[i] - mean; var += d * d; }
#pragma unroll
  for (int o = 16; o > 0; o >>= 1) var += __shfl_xor(var, o, 32);
  const float rstd = rsqrtf(var * (1.0f / D_) + 1e-5f);
  __bf16* yr = y + (size_t)row * D_;
#pragma unroll
  for (int i = 0; i < 16; ++i) {
    const int c = lane + 32 * i;
    yr[c] = (__bf16)((v[i] - mean) * rstd * g[c] + bta[c]);
  }
}

// ---------------- per-head L2 normalize (f32 -> bf16), one wave per 64-elem head ----------------
__global__ __launch_bounds__(256)
void l2norm_kernel(const float* __restrict__ x, __bf16* __restrict__ y, int units) {
  const int u = blockIdx.x * 8 + (threadIdx.x >> 5);
  const int lane = threadIdx.x & 31;
  if (u >= units) return;
  const float* xr = x + (size_t)u * DH_;
  const float a = xr[lane], b = xr[lane + 32];
  float s = a * a + b * b;
#pragma unroll
  for (int o = 16; o > 0; o >>= 1) s += __shfl_xor(s, o, 32);
  const float inv = 1.0f / fmaxf(sqrtf(s), 1e-12f);
  __bf16* yr = y + (size_t)u * DH_;
  yr[lane]      = (__bf16)(a * inv);
  yr[lane + 32] = (__bf16)(b * inv);
}

// ---------------- masked softmax + eps-smoothing (f32 logits -> bf16 probs) ----------------
__global__ __launch_bounds__(256)
void softmax_kernel(const float* __restrict__ logits, const unsigned char* __restrict__ pad,
                    __bf16* __restrict__ attn, int Lq, int Lk) {
  __shared__ float sred[256];
  const int row = blockIdx.x;                 // (b*H + h)*Lq + q
  const int b = row / (H_ * Lq);
  const float* lr = logits + (size_t)row * Lk;
  const unsigned char* pr = pad + (size_t)b * Lk;
  const int t = threadIdx.x;

  float mx = -3.0e38f;
  for (int k = t; k < Lk; k += 256) if (!pr[k]) mx = fmaxf(mx, lr[k]);
  sred[t] = mx; __syncthreads();
  for (int s = 128; s > 0; s >>= 1) { if (t < s) sred[t] = fmaxf(sred[t], sred[t + s]); __syncthreads(); }
  mx = sred[0]; __syncthreads();

  float sum = 0.f, cnt = 0.f;
  for (int k = t; k < Lk; k += 256) if (!pr[k]) { sum += expf(lr[k] - mx); cnt += 1.f; }
  sred[t] = sum; __syncthreads();
  for (int s = 128; s > 0; s >>= 1) { if (t < s) sred[t] += sred[t + s]; __syncthreads(); }
  sum = sred[0]; __syncthreads();
  sred[t] = cnt; __syncthreads();
  for (int s = 128; s > 0; s >>= 1) { if (t < s) sred[t] += sred[t + s]; __syncthreads(); }
  cnt = sred[0]; __syncthreads();

  const float inv = (1.0f - EPS_SM) / fmaxf(sum, 1e-8f);
  const float epsv = EPS_SM / fmaxf(cnt, 1.0f);
  __bf16* ar = attn + (size_t)row * Lk;
  for (int k = t; k < Lk; k += 256)
    ar[k] = pr[k] ? (__bf16)0.0f : (__bf16)(expf(lr[k] - mx) * inv + epsv);
}

// ---------------- f32 -> bf16 conversion (weights) ----------------
__global__ __launch_bounds__(256)
void convert_bf16_kernel(const float* __restrict__ x, __bf16* __restrict__ y, size_t n) {
  size_t i = (size_t)blockIdx.x * 256 + threadIdx.x;
  const size_t stride = (size_t)gridDim.x * 256;
  for (; i < n; i += stride) y[i] = (__bf16)x[i];
}

// ================= host orchestration =================
static void launch_gemm(int epi, const __bf16* A, const __bf16* W, const float* bias,
                        void* C, int M, int N, int K, int lda, int ldb, int ldc,
                        long long sAb, long long sAh, long long sBb, long long sBh,
                        long long sCb, long long sCh, int Hdiv, int Z,
                        float alpha, int LkVT, hipStream_t s) {
  dim3 grid(N / 64, M / 128, Z), blk(128);
  char* Cc = (char*)C;
  switch (epi) {
    case EPI_F32:   gemm_nt_kernel<EPI_F32>  <<<grid, blk, 0, s>>>(A, W, bias, Cc, K, lda, ldb, ldc, sAb, sAh, sBb, sBh, sCb, sCh, Hdiv, alpha, LkVT); break;
    case EPI_BF16:  gemm_nt_kernel<EPI_BF16> <<<grid, blk, 0, s>>>(A, W, bias, Cc, K, lda, ldb, ldc, sAb, sAh, sBb, sBh, sCb, sCh, Hdiv, alpha, LkVT); break;
    case EPI_GELU:  gemm_nt_kernel<EPI_GELU> <<<grid, blk, 0, s>>>(A, W, bias, Cc, K, lda, ldb, ldc, sAb, sAh, sBb, sBh, sCb, sCh, Hdiv, alpha, LkVT); break;
    case EPI_VT:    gemm_nt_kernel<EPI_VT>   <<<grid, blk, 0, s>>>(A, W, bias, Cc, K, lda, ldb, ldc, sAb, sAh, sBb, sBh, sCb, sCh, Hdiv, alpha, LkVT); break;
    default:        gemm_nt_kernel<EPI_RESID><<<grid, blk, 0, s>>>(A, W, bias, Cc, K, lda, ldb, ldc, sAb, sAh, sBb, sBh, sCb, sCh, Hdiv, alpha, LkVT); break;
  }
}

static void cross_attn(hipStream_t s,
                       float* resid_q, const float* lnq_g, const float* lnq_b,
                       float* resid_kv, const float* lnkv_g, const float* lnkv_b,
                       int Lq, int Lk, const unsigned char* pad,
                       const __bf16* wq, const float* bq, const __bf16* wk, const float* bk,
                       const __bf16* wv, const float* bv, const __bf16* wo, const float* bo,
                       __bf16* lnq_bf, __bf16* lnkv_bf, float* tmpf,
                       __bf16* q_bf, __bf16* k_bf, __bf16* vT_bf,
                       float* logits, __bf16* attn, __bf16* ctx) {
  const int Mq = B_ * Lq, Mk = B_ * Lk;
  layernorm_kernel<<<(Mq + 7) / 8, 256, 0, s>>>(resid_q, lnq_g, lnq_b, lnq_bf, Mq);
  layernorm_kernel<<<(Mk + 7) / 8, 256, 0, s>>>(resid_kv, lnkv_g, lnkv_b, lnkv_bf, Mk);
  // q/k projections -> f32, then per-head l2norm -> bf16
  launch_gemm(EPI_F32, lnq_bf, wq, bq, tmpf, Mq, D_, D_, D_, D_, D_, 0,0,0,0,0,0, 1, 1, 1.0f, 0, s);
  l2norm_kernel<<<(Mq * H_ + 7) / 8, 256, 0, s>>>(tmpf, q_bf, Mq * H_);
  launch_gemm(EPI_F32, lnkv_bf, wk, bk, tmpf, Mk, D_, D_, D_, D_, D_, 0,0,0,0,0,0, 1, 1, 1.0f, 0, s);
  l2norm_kernel<<<(Mk * H_ + 7) / 8, 256, 0, s>>>(tmpf, k_bf, Mk * H_);
  // v projection written transposed: vT[b][h][d][k]
  launch_gemm(EPI_VT, lnkv_bf, wv, bv, vT_bf, Mk, D_, D_, D_, D_, 0, 0,0,0,0,0,0, 1, 1, 1.0f, Lk, s);
  // logits[b][h][q][k] = SCALE * q . k   (batched over b*h, K=DH)
  launch_gemm(EPI_F32, q_bf, k_bf, nullptr, logits, Lq, Lk, DH_, D_, D_, Lk,
              (long long)Lq * D_, DH_, (long long)Lk * D_, DH_,
              (long long)H_ * Lq * Lk, (long long)Lq * Lk, H_, B_ * H_, SCALE_, 0, s);
  softmax_kernel<<<B_ * H_ * Lq, 256, 0, s>>>(logits, pad, attn, Lq, Lk);
  // ctx[b][q][h*64+d] = attn . vT  (batched, K=Lk)
  launch_gemm(EPI_BF16, attn, vT_bf, nullptr, ctx, Lq, DH_, Lk, Lk, Lk, D_,
              (long long)H_ * Lq * Lk, (long long)Lq * Lk,
              (long long)H_ * DH_ * Lk, (long long)DH_ * Lk,
              (long long)Lq * D_, DH_, H_, B_ * H_, 1.0f, 0, s);
  // o projection + residual accumulate into resid_q
  launch_gemm(EPI_RESID, ctx, wo, bo, resid_q, Mq, D_, D_, D_, D_, D_, 0,0,0,0,0,0, 1, 1, 1.0f, 0, s);
}

static void ffn_block(hipStream_t s, float* resid, const float* g, const float* bta,
                      const __bf16* w1, const float* b1, const __bf16* w2, const float* b2,
                      int Mrows, __bf16* ln_bf, __bf16* hid) {
  layernorm_kernel<<<(Mrows + 7) / 8, 256, 0, s>>>(resid, g, bta, ln_bf, Mrows);
  launch_gemm(EPI_GELU, ln_bf, w1, b1, hid, Mrows, DFF_, D_, D_, D_, DFF_, 0,0,0,0,0,0, 1, 1, 1.0f, 0, s);
  launch_gemm(EPI_RESID, hid, w2, b2, resid, Mrows, D_, DFF_, DFF_, DFF_, D_, 0,0,0,0,0,0, 1, 1, 1.0f, 0, s);
}

extern "C" void kernel_launch(void* const* d_in, const int* in_sizes, int n_in,
                              void* d_out, int out_size, void* d_ws, size_t ws_size,
                              hipStream_t stream) {
  (void)in_sizes; (void)n_in; (void)out_size; (void)ws_size;
  // ---- inputs (setup_inputs dict order) ----
  const float* in_lh = (const float*)d_in[0];
  const float* in_ph = (const float*)d_in[1];
  const unsigned char* l_pad = (const unsigned char*)d_in[2];
  const unsigned char* p_pad = (const unsigned char*)d_in[3];
  // LN params: [4..15] = {ln_l_q, ln_l_kv, ln_p_q, ln_p_kv, ln_l_ffn, ln_p_ffn} x {g,b}
  // attn:      [16..31] = lp_{wq,bq,wk,bk,wv,bv,wo,bo}, pl_{...}
  // ffn:       [32..39] = ffl_{w1,b1,w2,b2}, ffp_{...}
  auto F = [&](int i) { return (const float*)d_in[i]; };

  float* out_l = (float*)d_out;
  float* out_p = out_l + (size_t)ML_ * D_;

  // ---- workspace layout ----
  char* ws = (char*)d_ws;
  size_t off = 0;
  auto bump = [&](size_t bytes) -> char* {
    char* p = ws + off; off += (bytes + 255) & ~(size_t)255; return p;
  };
  const size_t AW = (size_t)D_ * D_;       // 262144 elems / layer
  const size_t FW = (size_t)DFF_ * D_;     // 1048576 elems / layer
  const size_t ASZ = NL_ * AW, FSZ = NL_ * FW;
  __bf16* wbf    = (__bf16*)bump((8 * ASZ + 4 * FSZ) * 2);
  __bf16* ln_l   = (__bf16*)bump((size_t)ML_ * D_ * 2);
  __bf16* ln_p   = (__bf16*)bump((size_t)MP_ * D_ * 2);
  float*  tmpf   = (float*) bump((size_t)MP_ * D_ * 4);
  __bf16* q_bf   = (__bf16*)bump((size_t)MP_ * D_ * 2);
  __bf16* k_bf   = (__bf16*)bump((size_t)MP_ * D_ * 2);
  __bf16* vT_bf  = (__bf16*)bump((size_t)MP_ * D_ * 2);
  float*  logits = (float*) bump((size_t)B_ * H_ * LL_ * LP_ * 4);
  __bf16* attn   = (__bf16*)bump((size_t)B_ * H_ * LL_ * LP_ * 2);
  __bf16* ctx    = (__bf16*)bump((size_t)MP_ * D_ * 2);
  __bf16* hid    = (__bf16*)logits;  // FFN hidden aliases logits (disjoint lifetimes)

  // ---- residual streams live in d_out ----
  hipMemcpyAsync(out_l, in_lh, (size_t)ML_ * D_ * 4, hipMemcpyDeviceToDevice, stream);
  hipMemcpyAsync(out_p, in_ph, (size_t)MP_ * D_ * 4, hipMemcpyDeviceToDevice, stream);

  // ---- convert all weights to bf16 (weights: 48MB bf16, resident in L2) ----
  const int widx[12]  = {16, 18, 20, 22, 24, 26, 28, 30, 32, 34, 36, 38};
  for (int j = 0; j < 12; ++j) {
    const size_t cnt = (j < 8) ? ASZ : FSZ;
    const size_t wo  = (j < 8) ? (size_t)j * ASZ : 8 * ASZ + (size_t)(j - 8) * FSZ;
    convert_bf16_kernel<<<2048, 256, 0, stream>>>(F(widx[j]), wbf + wo, cnt);
  }
  __bf16* w_lp[4], * w_pl[4];
  for (int j = 0; j < 4; ++j) { w_lp[j] = wbf + (size_t)j * ASZ; w_pl[j] = wbf + (size_t)(4 + j) * ASZ; }
  __bf16* w_ffl1 = wbf + 8 * ASZ;
  __bf16* w_ffl2 = w_ffl1 + FSZ;
  __bf16* w_ffp1 = w_ffl2 + FSZ;
  __bf16* w_ffp2 = w_ffp1 + FSZ;

  for (int i = 0; i < NL_; ++i) {
    const size_t dO = (size_t)i * D_, fO = (size_t)i * DFF_;
    // ---- direction 1: l attends to p ----
    cross_attn(stream,
               out_l, F(4) + dO,  F(5) + dO,          // ln_l_q
               out_p, F(10) + dO, F(11) + dO,         // ln_p_kv
               LL_, LP_, p_pad,
               w_lp[0] + i * AW, F(17) + dO, w_lp[1] + i * AW, F(19) + dO,
               w_lp[2] + i * AW, F(21) + dO, w_lp[3] + i * AW, F(23) + dO,
               ln_l, ln_p, tmpf, q_bf, k_bf, vT_bf, logits, attn, ctx);
    ffn_block(stream, out_l, F(12) + dO, F(13) + dO,  // ln_l_ffn
              w_ffl1 + i * FW, F(33) + fO, w_ffl2 + i * FW, F(35) + dO, ML_, ln_l, hid);
    // ---- direction 2: p attends to (updated) l ----
    cross_attn(stream,
               out_p, F(8) + dO, F(9) + dO,           // ln_p_q
               out_l, F(6) + dO, F(7) + dO,           // ln_l_kv
               LP_, LL_, l_pad,
               w_pl[0] + i * AW, F(25) + dO, w_pl[1] + i * AW, F(27) + dO,
               w_pl[2] + i * AW, F(29) + dO, w_pl[3] + i * AW, F(31) + dO,
               ln_p, ln_l, tmpf, q_bf, k_bf, vT_bf, logits, attn, ctx);
    ffn_block(stream, out_p, F(14) + dO, F(15) + dO,  // ln_p_ffn
              w_ffp1 + i * FW, F(37) + fO, w_ffp2 + i * FW, F(39) + dO, MP_, ln_p, hid);
  }
}
